// LatentODE_27754078667422
// MI455X (gfx1250) — matrix operand order repaired and checked by reading, not compiled, not run on hardware
//
#include <hip/hip_runtime.h>
#include <hip/hip_bf16.h>

typedef _Float16 h16;
typedef __attribute__((ext_vector_type(16))) _Float16 v16h;
typedef __attribute__((ext_vector_type(8)))  _Float16 v8h;
typedef __attribute__((ext_vector_type(8)))  float    v8f;

#define B_      8192
#define S_      128
#define T_      64
#define OBS_    64
#define LAT_    128
#define SUBST_  4

#define ACT_NONE 0
#define ACT_TANH 1
#define ACT_RELU 2
#define OUT_H16     0
#define OUT_F32_LDS 1
#define OUT_F32_G   2

// One 16-row GEMM tile:  D[16 x N] = act(A[16 x K] * Bt^T + bias)
// A: LDS, row-major f16, row stride `astride` halves (multiple of 8).
// Bt: global f16, N-major (Bt[n*kpad + k]) so each lane's B fragment is a
//     contiguous 32B load holding k = 16*hi + s, the same k order as A.
// 4 waves split N; NTPW = (N/16)/4 column tiles per wave.
template<int KTILES, int NTPW, int ACT, int OUTM>
__device__ __forceinline__ void gemm16(const h16* A, int astride,
                                       const h16* Bt, int kpad,
                                       const float* bias,
                                       h16* outH, float* outF, int ostride,
                                       int wave, int lane)
{
    const int rlo = lane & 15;
    const int hi  = lane >> 4;          // lanes 16-31 hold the +16 K half
    const h16* ap = A + rlo * astride + hi * 16;

    v8f zero = {};
    v8f acc[NTPW];
#pragma unroll
    for (int t = 0; t < NTPW; ++t) acc[t] = zero;

#pragma unroll
    for (int k = 0; k < KTILES; ++k) {
        // A fragment: lane holds row rlo, K = k*32 + hi*16 + {0..15}
        const h16* abase = ap + k * 32;
        v8h alo = *(const v8h*)(abase);
        v8h ahi = *(const v8h*)(abase + 8);
        v16h a;
#pragma unroll
        for (int j = 0; j < 8; ++j) { a[j] = alo[j]; a[8 + j] = ahi[j]; }
#pragma unroll
        for (int t = 0; t < NTPW; ++t) {
            const int ntile = wave * NTPW + t;
            const h16* bbase = Bt + (size_t)(ntile * 16 + rlo) * kpad + k * 32 + hi * 16;
            v16h b = *(const v16h*)(bbase);   // 32B contiguous: K k*32+hi*16 .. +15
            acc[t] = __builtin_amdgcn_wmma_f32_16x16x32_f16(
                         false, a, false, b, (short)0, acc[t], false, false);
        }
    }
    __syncthreads();   // all A reads complete before any (possibly aliasing) writes
#pragma unroll
    for (int t = 0; t < NTPW; ++t) {
        const int col = (wave * NTPW + t) * 16 + rlo;
        const float bv = bias[col];
#pragma unroll
        for (int r = 0; r < 8; ++r) {
            const int row = r + hi * 8;     // C/D layout: lanes16-31 -> rows 8-15
            float v = acc[t][r] + bv;
            if (ACT == ACT_TANH)      v = tanhf(v);
            else if (ACT == ACT_RELU) v = fmaxf(v, 0.0f);
            if (OUTM == OUT_H16)      outH[row * ostride + col] = (h16)v;
            else                      outF[(size_t)row * ostride + col] = v;
        }
    }
}

__device__ __forceinline__ void ode_f(const h16* in, h16* hid, float* kb,
                                      const h16* bt1, const float* b1,
                                      const h16* bt2, const float* b2,
                                      int wave, int lane)
{
    gemm16<4, 4, ACT_TANH, OUT_H16>(in, 128, bt1, 128, b1, hid, nullptr, 256, wave, lane);
    __syncthreads();
    gemm16<8, 2, ACT_NONE, OUT_F32_LDS>(hid, 256, bt2, 256, b2, nullptr, kb, 128, wave, lane);
    __syncthreads();
}

__device__ __forceinline__ void decode_step(const h16* z, h16* hid,
                                            const h16* btd1, const float* bd1,
                                            const h16* btd2, const float* bd2,
                                            float* predt, int wave, int lane)
{
    gemm16<4, 4, ACT_RELU, OUT_H16>(z, 128, btd1, 128, bd1, hid, nullptr, 256, wave, lane);
    __syncthreads();
    gemm16<8, 1, ACT_NONE, OUT_F32_G>(hid, 256, btd2, 256, bd2, nullptr, predt, 64, wave, lane);
    __syncthreads();
}

__global__ void __launch_bounds__(128)
latent_ode_kernel(const float* __restrict__ obs,
                  const float* __restrict__ pred_times,
                  const float* __restrict__ eps,
                  const h16* bt_i2h, const float* b_i2h,
                  const h16* bt_h2o, const float* b_h2o,
                  const h16* bt_ode1, const float* b_ode1,
                  const h16* bt_ode2, const float* b_ode2,
                  const h16* bt_dec1, const float* b_dec1,
                  const h16* bt_dec2, const float* b_dec2,
                  float* __restrict__ out)
{
    __shared__ __align__(32) char smem[59392];
    h16*   combA = (h16*)(smem);            // 16x320 f16 (encoder [x|h])
    float* stg   = (float*)(smem + 10240);  // 16x256 f32 (latent head out / RK4 k)
    float* zf32  = (float*)(smem + 26624);  // 16x128 f32 (current z)
    float* zacc  = (float*)(smem + 34816);  // 16x128 f32 (RK4 accumulator)
    h16*   zf16  = (h16*)(smem + 43008);    // 16x128 f16
    h16*   ftb   = (h16*)(smem + 47104);    // 16x128 f16 (RK4 stage input)
    h16*   hid   = (h16*)(smem + 51200);    // 16x256 f16 (MLP hidden)

    const int tid  = threadIdx.x;
    const int wave = tid >> 5;
    const int lane = tid & 31;
    const int r0   = blockIdx.x * 16;       // this block's 16 batch rows

    // ---- Phase 1: encoder RNN (reversed sequence), h kept in LDS ----
    for (int i = tid; i < 16 * 320; i += 128) combA[i] = (h16)0.0f;
    __syncthreads();

    for (int s = S_ - 1; s >= 0; --s) {
        for (int i = tid; i < 16 * 64; i += 128) {
            const int row = i >> 6, col = i & 63;
            const float xv = obs[((size_t)(r0 + row) * S_ + s) * OBS_ + col];
            combA[row * 320 + col] = (h16)xv;
        }
        __syncthreads();
        // h_new = tanh([x|h] @ Wi2h + b), written back into combA cols 64..319
        gemm16<10, 4, ACT_TANH, OUT_H16>(combA, 320, bt_i2h, 320, b_i2h,
                                         combA + 64, nullptr, 320, wave, lane);
        __syncthreads();
    }

    // ---- Phase 2: latent head + reparameterization ----
    gemm16<8, 4, ACT_NONE, OUT_F32_LDS>(combA + 64, 320, bt_h2o, 256, b_h2o,
                                        nullptr, stg, 256, wave, lane);
    __syncthreads();
    const size_t mean_off = (size_t)T_ * B_ * OBS_;
    const size_t logv_off = mean_off + (size_t)B_ * LAT_;
    for (int i = tid; i < 16 * 128; i += 128) {
        const int row = i >> 7, col = i & 127;
        const size_t gr = (size_t)(r0 + row);
        const float m  = stg[row * 256 + col];
        const float lv = stg[row * 256 + 128 + col];
        out[mean_off + gr * LAT_ + col] = m;
        out[logv_off + gr * LAT_ + col] = lv;
        const float z = m + eps[gr * LAT_ + col] * expf(0.5f * lv);
        zf32[i] = z;
        zf16[i] = (h16)z;
    }
    __syncthreads();

    // ---- Phase 3: decode z0, then RK4 ODE integration + decode per interval ----
    decode_step(zf16, hid, bt_dec1, b_dec1, bt_dec2, b_dec2,
                out + (size_t)r0 * OBS_, wave, lane);

    for (int iv = 0; iv < T_ - 1; ++iv) {
        const float dt = (pred_times[iv + 1] - pred_times[iv]) / (float)SUBST_;
        for (int ss = 0; ss < SUBST_; ++ss) {
            // k1
            ode_f(zf16, hid, stg, bt_ode1, b_ode1, bt_ode2, b_ode2, wave, lane);
            for (int i = tid; i < 2048; i += 128) {
                const float k = stg[i], z = zf32[i];
                zacc[i] = z + (dt * (1.0f / 6.0f)) * k;
                ftb[i]  = (h16)(z + 0.5f * dt * k);
            }
            __syncthreads();
            // k2
            ode_f(ftb, hid, stg, bt_ode1, b_ode1, bt_ode2, b_ode2, wave, lane);
            for (int i = tid; i < 2048; i += 128) {
                const float k = stg[i], z = zf32[i];
                zacc[i] += (dt * (1.0f / 3.0f)) * k;
                ftb[i]   = (h16)(z + 0.5f * dt * k);
            }
            __syncthreads();
            // k3
            ode_f(ftb, hid, stg, bt_ode1, b_ode1, bt_ode2, b_ode2, wave, lane);
            for (int i = tid; i < 2048; i += 128) {
                const float k = stg[i], z = zf32[i];
                zacc[i] += (dt * (1.0f / 3.0f)) * k;
                ftb[i]   = (h16)(z + dt * k);
            }
            __syncthreads();
            // k4 + advance
            ode_f(ftb, hid, stg, bt_ode1, b_ode1, bt_ode2, b_ode2, wave, lane);
            for (int i = tid; i < 2048; i += 128) {
                const float k  = stg[i];
                const float zn = zacc[i] + (dt * (1.0f / 6.0f)) * k;
                zf32[i] = zn;
                zf16[i] = (h16)zn;
            }
            __syncthreads();
        }
        decode_step(zf16, hid, bt_dec1, b_dec1, bt_dec2, b_dec2,
                    out + ((size_t)(iv + 1) * B_ + r0) * OBS_, wave, lane);
    }
}

// Pre-transpose a (K x N) f32 weight into N-major f16: Bt[n*K + k] = W[k*N + n]
__global__ void transpose_f16(const float* __restrict__ W, h16* __restrict__ Bt,
                              int K, int N)
{
    const int idx = blockIdx.x * blockDim.x + threadIdx.x;
    if (idx < K * N) {
        const int n = idx / K;
        const int k = idx - n * K;
        Bt[(size_t)n * K + k] = (h16)W[(size_t)k * N + n];
    }
}

extern "C" void kernel_launch(void* const* d_in, const int* in_sizes, int n_in,
                              void* d_out, int out_size, void* d_ws, size_t ws_size,
                              hipStream_t stream)
{
    (void)in_sizes; (void)n_in; (void)out_size; (void)ws_size;
    const float* obs        = (const float*)d_in[0];
    /* d_in[1] = observed_times: unused by the reference computation */
    const float* pred_times = (const float*)d_in[2];
    const float* eps        = (const float*)d_in[3];
    const float* Wi2h  = (const float*)d_in[4];  const float* bi2h  = (const float*)d_in[5];
    const float* Wh2o  = (const float*)d_in[6];  const float* bh2o  = (const float*)d_in[7];
    const float* Wode1 = (const float*)d_in[8];  const float* bode1 = (const float*)d_in[9];
    const float* Wode2 = (const float*)d_in[10]; const float* bode2 = (const float*)d_in[11];
    const float* Wdec1 = (const float*)d_in[12]; const float* bdec1 = (const float*)d_in[13];
    const float* Wdec2 = (const float*)d_in[14]; const float* bdec2 = (const float*)d_in[15];

    // f16 transposed weight copies in scratch (512 KB total)
    h16* ws      = (h16*)d_ws;
    h16* bt_i2h  = ws;             // 256 x 320
    h16* bt_h2o  = ws + 81920;     // 256 x 256
    h16* bt_ode1 = ws + 147456;    // 256 x 128
    h16* bt_ode2 = ws + 180224;    // 128 x 256
    h16* bt_dec1 = ws + 212992;    // 256 x 128
    h16* bt_dec2 = ws + 245760;    //  64 x 256

    auto tr = [&](const float* W, h16* Bt, int K, int N) {
        const int total = K * N;
        transpose_f16<<<(total + 255) / 256, 256, 0, stream>>>(W, Bt, K, N);
    };
    tr(Wi2h,  bt_i2h,  320, 256);
    tr(Wh2o,  bt_h2o,  256, 256);
    tr(Wode1, bt_ode1, 128, 256);
    tr(Wode2, bt_ode2, 256, 128);
    tr(Wdec1, bt_dec1, 128, 256);
    tr(Wdec2, bt_dec2, 256, 64);

    latent_ode_kernel<<<B_ / 16, 128, 0, stream>>>(
        obs, pred_times, eps,
        bt_i2h, bi2h, bt_h2o, bh2o,
        bt_ode1, bode1, bt_ode2, bode2,
        bt_dec1, bdec1, bt_dec2, bdec2,
        (float*)d_out);
}
